// DecoderOnlyTransformer_63599875719301
// MI455X (gfx1250) — matrix operand order repaired and checked by reading, not compiled, not run on hardware
//
#include <hip/hip_runtime.h>

// ---------------------------------------------------------------------------
// Types
// ---------------------------------------------------------------------------
typedef __bf16 bf16;
typedef bf16  v8bf  __attribute__((ext_vector_type(8)));
typedef bf16  v16bf __attribute__((ext_vector_type(16)));
typedef float v8f   __attribute__((ext_vector_type(8)));

#define TSEQ   128
#define DMODEL 512
#define NHEADS 8
#define DHEAD  64
#define NLAYERS 4
#define DFF    2048
#define THIST  120
#define NSTEPS 8
#define DIN    32
#define NSUB   5
#define NBATCH 4
#define EPSV   1e-5f

#define WMMA_BF16(A_, B_, C_)                                                 \
  __builtin_amdgcn_wmma_f32_16x16x32_bf16(false, (A_), false, (B_), (short)0, \
                                          (C_), false, false)

// ---------------------------------------------------------------------------
// WMMA fragment loaders (bf16, 16x16x32, wave32 layouts per CDNA5 ISA 7.12.2)
// A (16x32, row-major): lane m=l&15, kb=(l&16)?8:0; elems 0..7=K kb..kb+7,
//   elems 8..15=K 16+kb..16+kb+7  -> two contiguous 16B chunks.
// B (32x16): stored as W[N][K] row-major (B = W^T; column n = W row n):
//   lane n=l&15, kb=(l&16)?8:0; elems 0..7=K kb..kb+7, elems 8..15=K 16+kb..16+kb+7.
// ---------------------------------------------------------------------------
__device__ __forceinline__ v16bf load_a_frag(const bf16* base, int ld) {
  const int lane = threadIdx.x & 31;
  const bf16* p  = base + (lane & 15) * ld + ((lane & 16) >> 1);
  v8bf lo = *reinterpret_cast<const v8bf*>(p);
  v8bf hi = *reinterpret_cast<const v8bf*>(p + 16);
  return __builtin_shufflevector(lo, hi, 0, 1, 2, 3, 4, 5, 6, 7,
                                 8, 9, 10, 11, 12, 13, 14, 15);
}

__device__ __forceinline__ v16bf load_b_frag(const bf16* w, int ld) {
  const int lane = threadIdx.x & 31;
  const bf16* p  = w + (lane & 15) * ld + ((lane & 16) >> 1);
  v8bf lo = *reinterpret_cast<const v8bf*>(p);
  v8bf hi = *reinterpret_cast<const v8bf*>(p + 16);
  return __builtin_shufflevector(lo, hi, 0, 1, 2, 3, 4, 5, 6, 7,
                                 8, 9, 10, 11, 12, 13, 14, 15);
}

// ---------------------------------------------------------------------------
// Wave-level GEMM strips with *named* accumulators (no arrays -> no allocas,
// accumulators stay in VGPRs across the K loop).
// ---------------------------------------------------------------------------
struct Acc4 { v8f c0, c1, c2, c3; };
struct Acc8 { v8f c0, c1, c2, c3, c4, c5, c6, c7; };

__device__ __forceinline__ Acc4 gemm_strip4(const bf16* A, int lda,
                                            const bf16* W, int ldw, int K) {
  const v8f z = {0.f, 0.f, 0.f, 0.f, 0.f, 0.f, 0.f, 0.f};
  v8f c0 = z, c1 = z, c2 = z, c3 = z;
  for (int k = 0; k < K; k += 32) {
    v16bf a = load_a_frag(A + k, lda);
    c0 = WMMA_BF16(a, load_b_frag(W + 0  * ldw + k, ldw), c0);
    c1 = WMMA_BF16(a, load_b_frag(W + 16 * ldw + k, ldw), c1);
    c2 = WMMA_BF16(a, load_b_frag(W + 32 * ldw + k, ldw), c2);
    c3 = WMMA_BF16(a, load_b_frag(W + 48 * ldw + k, ldw), c3);
  }
  Acc4 r = {c0, c1, c2, c3};
  return r;
}

__device__ __forceinline__ Acc8 gemm_strip8(const bf16* A, int lda,
                                            const bf16* W, int ldw, int K) {
  const v8f z = {0.f, 0.f, 0.f, 0.f, 0.f, 0.f, 0.f, 0.f};
  v8f c0 = z, c1 = z, c2 = z, c3 = z, c4 = z, c5 = z, c6 = z, c7 = z;
  for (int k = 0; k < K; k += 32) {
    v16bf a = load_a_frag(A + k, lda);
    c0 = WMMA_BF16(a, load_b_frag(W + 0   * ldw + k, ldw), c0);
    c1 = WMMA_BF16(a, load_b_frag(W + 16  * ldw + k, ldw), c1);
    c2 = WMMA_BF16(a, load_b_frag(W + 32  * ldw + k, ldw), c2);
    c3 = WMMA_BF16(a, load_b_frag(W + 48  * ldw + k, ldw), c3);
    c4 = WMMA_BF16(a, load_b_frag(W + 64  * ldw + k, ldw), c4);
    c5 = WMMA_BF16(a, load_b_frag(W + 80  * ldw + k, ldw), c5);
    c6 = WMMA_BF16(a, load_b_frag(W + 96  * ldw + k, ldw), c6);
    c7 = WMMA_BF16(a, load_b_frag(W + 112 * ldw + k, ldw), c7);
  }
  Acc8 r = {c0, c1, c2, c3, c4, c5, c6, c7};
  return r;
}

// ---------------------------------------------------------------------------
// Epilogue helpers: v8f by value; only #pragma unroll vector subscripts.
// C tile layout: element g -> row = row0 + g + rh, col = col0 + cn
// ---------------------------------------------------------------------------
__device__ __forceinline__ float gelu_tanh(float x) {
  const float c = 0.7978845608028654f;  // sqrt(2/pi)
  float x3 = x * x * x;
  return 0.5f * x * (1.0f + tanhf(c * (x + 0.044715f * x3)));
}

__device__ __forceinline__ void st_scale(v8f c, bf16* dst, int ld, int row0,
                                         int col0, int rh, int cn, float s) {
#pragma unroll
  for (int g = 0; g < 8; ++g)
    dst[(row0 + g + rh) * ld + col0 + cn] = (bf16)(c[g] * s);
}

__device__ __forceinline__ void st_transposed(v8f c, bf16* dst, int ldT,
                                              int row0, int col0, int rh,
                                              int cn) {
#pragma unroll
  for (int g = 0; g < 8; ++g)
    dst[(col0 + cn) * ldT + row0 + g + rh] = (bf16)c[g];
}

__device__ __forceinline__ void st_embed(v8f c, bf16* hb, const float* b_emb,
                                         int row0, int col0, int rh, int cn,
                                         float scale) {
  const int col = col0 + cn;
  const float dv =
      __expf((-9.210340371976184f / (float)DMODEL) * (float)(2 * (col >> 1)));
  const float be = b_emb[col];
  const int odd = col & 1;
#pragma unroll
  for (int g = 0; g < 8; ++g) {
    const int row = row0 + g + rh;
    float x = (c[g] + be) * scale;
    const float ang = (float)row * dv;
    x += odd ? __cosf(ang) : __sinf(ang);
    hb[row * DMODEL + col] = (bf16)x;
  }
}

__device__ __forceinline__ void st_resid(v8f c, bf16* hb, int row0, int col0,
                                         int rh, int cn) {
#pragma unroll
  for (int g = 0; g < 8; ++g) {
    const int idx = (row0 + g + rh) * DMODEL + col0 + cn;
    hb[idx] = (bf16)(c[g] + (float)hb[idx]);
  }
}

__device__ __forceinline__ void st_resid_bias(v8f c, bf16* hb,
                                              const float* bias, int row0,
                                              int col0, int rh, int cn) {
  const int col = col0 + cn;
  const float bv = bias[col];
#pragma unroll
  for (int g = 0; g < 8; ++g) {
    const int idx = (row0 + g + rh) * DMODEL + col;
    hb[idx] = (bf16)(c[g] + bv + (float)hb[idx]);
  }
}

__device__ __forceinline__ void st_gelu(v8f c, bf16* m, const float* bias,
                                        int row0, int col0, int rh, int cn) {
  const int col = col0 + cn;
  const float bv = bias[col];
#pragma unroll
  for (int g = 0; g < 8; ++g)
    m[(row0 + g + rh) * DFF + col] = (bf16)gelu_tanh(c[g] + bv);
}

// In-place LayerNorm over rows of h[128][512] (bf16 in LDS). 2 threads/row.
__device__ __forceinline__ void layernorm_inplace(bf16* h, const float* g,
                                                  const float* bb, int tid) {
  const int row  = tid >> 1;
  const int half = tid & 1;
  bf16* hr = h + row * DMODEL + half * 256;
  float s = 0.f, s2 = 0.f;
  for (int i = 0; i < 256; ++i) { float x = (float)hr[i]; s += x; s2 += x * x; }
  s  += __shfl_xor(s, 1, 32);
  s2 += __shfl_xor(s2, 1, 32);
  const float mu  = s * (1.0f / DMODEL);
  const float var = s2 * (1.0f / DMODEL) - mu * mu;
  const float inv = rsqrtf(var + EPSV);
  for (int i = 0; i < 256; ++i) {
    const int col = half * 256 + i;
    float x = (float)hr[i];
    hr[i] = (bf16)(((x - mu) * inv) * g[col] + bb[col]);
  }
}

// ---------------------------------------------------------------------------
// fp32 -> bf16 weight conversion (grid-stride)
// ---------------------------------------------------------------------------
__global__ void f32_to_bf16_kernel(const float* __restrict__ src,
                                   bf16* __restrict__ dst, int n) {
  int i = blockIdx.x * blockDim.x + threadIdx.x;
  const int stride = gridDim.x * blockDim.x;
  for (; i < n; i += stride) dst[i] = (bf16)src[i];
}

// ---------------------------------------------------------------------------
// Persistent rollout kernel: one workgroup per batch element.
// 256 threads = 8 wave32; wave w owns row strip [16w, 16w+16).
// ---------------------------------------------------------------------------
__global__ __launch_bounds__(256)
void rollout_kernel(const float* __restrict__ history,
                    const float* __restrict__ b_emb,
                    const float* __restrict__ ln1_g, const float* __restrict__ ln1_b,
                    const float* __restrict__ b1,   const float* __restrict__ b2,
                    const float* __restrict__ ln2_g, const float* __restrict__ ln2_b,
                    const float* __restrict__ W_emb_f,
                    const float* __restrict__ Wr1,  const float* __restrict__ br1,
                    const float* __restrict__ Wr2,  const float* __restrict__ br2,
                    const bf16* __restrict__ wemb,  // [512][32]
                    const bf16* __restrict__ wq,    // [L][512][512]
                    const bf16* __restrict__ wk,
                    const bf16* __restrict__ wv,
                    const bf16* __restrict__ wo,
                    const bf16* __restrict__ w1,    // [L][2048][512]
                    const bf16* __restrict__ w2,    // [L][512][2048]
                    bf16* __restrict__ m_ws,        // [B][128][2048]
                    float* __restrict__ buf_ws,     // [B][128][32]
                    float* __restrict__ out)        // [B][8][32]
{
  __shared__ __align__(128) char smem[311296];     // 304 KB of 320 KB WGP LDS
  bf16* hb = (bf16*)(smem);                        // [128][512] activations
  bf16* tb = (bf16*)(smem + 131072);               // [128][512] attn concat
  bf16* qb = (bf16*)(smem + 262144);               // [128][64]
  bf16* kb = (bf16*)(smem + 278528);               // [128][64]
  bf16* vt = (bf16*)(smem + 294912);               // [64][128]  (V transposed)
  bf16* sb = (bf16*)(smem + 262144);               // [128][128] aliases qb+kb
  float* comb = (float*)(smem + 131072);           // [1024] refine (aliases tb)
  float* actv = (float*)(smem + 131072 + 4096);    // [2048]
  float* curv = (float*)(smem + 131072 + 12288);   // [32]

  const int b    = blockIdx.x;
  const int tid  = threadIdx.x;
  const int lane = tid & 31;
  const int wid  = tid >> 5;
  const int r0   = wid * 16;
  const int cn   = lane & 15;
  const int rh   = (lane & 16) >> 1;               // +8 row offset upper half

  float* buf  = buf_ws + b * TSEQ * DIN;
  bf16*  mloc = m_ws + (size_t)b * TSEQ * DFF;
  const float scale  = 22.627416997969522f;        // sqrt(512)
  const float qscale = 0.125f;                     // 1/sqrt(64)
  const float NEG    = -3.4028234663852886e38f;

  // init rollout buffer: [:120] = history, rest 0
  for (int idx = tid; idx < TSEQ * DIN; idx += 256) {
    const int t = idx / DIN, i = idx % DIN;
    buf[idx] = (t < THIST) ? history[(b * THIST + t) * DIN + i] : 0.0f;
  }
  __syncthreads();

  for (int step = 0; step < NSTEPS; ++step) {
    // -------- embedding: h = (buf @ Wemb^T + b)*scale + pe --------
    for (int idx = tid; idx < TSEQ * DIN; idx += 256) {
      const int t = idx / DIN, i = idx % DIN;
      qb[t * DHEAD + i] = (bf16)buf[idx];          // stage A bf16 (lda=64)
    }
    __syncthreads();
    for (int nc = 0; nc < DMODEL; nc += 64) {
      Acc4 r = gemm_strip4(qb + r0 * DHEAD, DHEAD, wemb + nc * DIN, DIN, DIN);
      st_embed(r.c0, hb, b_emb, r0, nc + 0,  rh, cn, scale);
      st_embed(r.c1, hb, b_emb, r0, nc + 16, rh, cn, scale);
      st_embed(r.c2, hb, b_emb, r0, nc + 32, rh, cn, scale);
      st_embed(r.c3, hb, b_emb, r0, nc + 48, rh, cn, scale);
    }
    __syncthreads();

    // -------- transformer layers --------
    for (int l = 0; l < NLAYERS; ++l) {
      for (int hd = 0; hd < NHEADS; ++hd) {
        const bf16* wq_h = wq + ((size_t)l * DMODEL + hd * DHEAD) * DMODEL;
        const bf16* wk_h = wk + ((size_t)l * DMODEL + hd * DHEAD) * DMODEL;
        const bf16* wv_h = wv + ((size_t)l * DMODEL + hd * DHEAD) * DMODEL;
        {  // Q (pre-scaled by 1/sqrt(DHEAD))
          Acc4 r = gemm_strip4(hb + r0 * DMODEL, DMODEL, wq_h, DMODEL, DMODEL);
          st_scale(r.c0, qb, DHEAD, r0, 0,  rh, cn, qscale);
          st_scale(r.c1, qb, DHEAD, r0, 16, rh, cn, qscale);
          st_scale(r.c2, qb, DHEAD, r0, 32, rh, cn, qscale);
          st_scale(r.c3, qb, DHEAD, r0, 48, rh, cn, qscale);
        }
        {  // K
          Acc4 r = gemm_strip4(hb + r0 * DMODEL, DMODEL, wk_h, DMODEL, DMODEL);
          st_scale(r.c0, kb, DHEAD, r0, 0,  rh, cn, 1.0f);
          st_scale(r.c1, kb, DHEAD, r0, 16, rh, cn, 1.0f);
          st_scale(r.c2, kb, DHEAD, r0, 32, rh, cn, 1.0f);
          st_scale(r.c3, kb, DHEAD, r0, 48, rh, cn, 1.0f);
        }
        {  // V stored transposed (so S@V B-columns stay contiguous)
          Acc4 r = gemm_strip4(hb + r0 * DMODEL, DMODEL, wv_h, DMODEL, DMODEL);
          st_transposed(r.c0, vt, TSEQ, r0, 0,  rh, cn);
          st_transposed(r.c1, vt, TSEQ, r0, 16, rh, cn);
          st_transposed(r.c2, vt, TSEQ, r0, 32, rh, cn);
          st_transposed(r.c3, vt, TSEQ, r0, 48, rh, cn);
        }
        __syncthreads();

        // scores S = Q K^T (scaled) -> raw logits to sb (after barrier: sb
        // aliases qb/kb, so all reads must retire first)
        Acc8 s = gemm_strip8(qb + r0 * DHEAD, DHEAD, kb, DHEAD, DHEAD);
        __syncthreads();
        st_scale(s.c0, sb, TSEQ, r0, 0,   rh, cn, 1.0f);
        st_scale(s.c1, sb, TSEQ, r0, 16,  rh, cn, 1.0f);
        st_scale(s.c2, sb, TSEQ, r0, 32,  rh, cn, 1.0f);
        st_scale(s.c3, sb, TSEQ, r0, 48,  rh, cn, 1.0f);
        st_scale(s.c4, sb, TSEQ, r0, 64,  rh, cn, 1.0f);
        st_scale(s.c5, sb, TSEQ, r0, 80,  rh, cn, 1.0f);
        st_scale(s.c6, sb, TSEQ, r0, 96,  rh, cn, 1.0f);
        st_scale(s.c7, sb, TSEQ, r0, 112, rh, cn, 1.0f);
        __syncthreads();

        // softmax pass over sb rows; reversed causal mask (attend where k>q,
        // faithful to reference; fully-masked rows degrade to uniform 1/128)
        {
          const int row  = tid >> 1;
          const int half = tid & 1;
          bf16* srow = sb + row * TSEQ + half * 64;
          float mx = NEG;
          for (int i = 0; i < 64; ++i) {
            const int col = half * 64 + i;
            const float x = (col > row) ? (float)srow[i] : NEG;
            mx = fmaxf(mx, x);
          }
          mx = fmaxf(mx, __shfl_xor(mx, 1, 32));
          float sum = 0.f;
          for (int i = 0; i < 64; ++i) {
            const int col = half * 64 + i;
            const float x = (col > row) ? (float)srow[i] : NEG;
            const float e = __expf(x - mx);
            srow[i] = (bf16)e;
            sum += e;
          }
          sum += __shfl_xor(sum, 1, 32);
          const float inv = 1.0f / sum;
          for (int i = 0; i < 64; ++i)
            srow[i] = (bf16)((float)srow[i] * inv);
        }
        __syncthreads();

        {  // attn = S @ V  -> concat slice of tb
          Acc4 r = gemm_strip4(sb + r0 * TSEQ, TSEQ, vt, TSEQ, TSEQ);
          const int cb = hd * DHEAD;
          st_scale(r.c0, tb, DMODEL, r0, cb + 0,  rh, cn, 1.0f);
          st_scale(r.c1, tb, DMODEL, r0, cb + 16, rh, cn, 1.0f);
          st_scale(r.c2, tb, DMODEL, r0, cb + 32, rh, cn, 1.0f);
          st_scale(r.c3, tb, DMODEL, r0, cb + 48, rh, cn, 1.0f);
        }
        __syncthreads();  // protect qb/kb/vt/sb for next head
      }

      // output projection + residual (in place on hb)
      for (int nc = 0; nc < DMODEL; nc += 64) {
        Acc4 r = gemm_strip4(tb + r0 * DMODEL, DMODEL,
                             wo + ((size_t)l * DMODEL + nc) * DMODEL, DMODEL,
                             DMODEL);
        st_resid(r.c0, hb, r0, nc + 0,  rh, cn);
        st_resid(r.c1, hb, r0, nc + 16, rh, cn);
        st_resid(r.c2, hb, r0, nc + 32, rh, cn);
        st_resid(r.c3, hb, r0, nc + 48, rh, cn);
      }
      __syncthreads();
      layernorm_inplace(hb, ln1_g + l * DMODEL, ln1_b + l * DMODEL, tid);
      __syncthreads();

      // FFN pass 1: m = gelu(h @ W1^T + b1) -> bf16 in L2-resident scratch
      for (int nc = 0; nc < DFF; nc += 64) {
        Acc4 r = gemm_strip4(hb + r0 * DMODEL, DMODEL,
                             w1 + ((size_t)l * DFF + nc) * DMODEL, DMODEL,
                             DMODEL);
        const float* bb = b1 + l * DFF;
        st_gelu(r.c0, mloc, bb, r0, nc + 0,  rh, cn);
        st_gelu(r.c1, mloc, bb, r0, nc + 16, rh, cn);
        st_gelu(r.c2, mloc, bb, r0, nc + 32, rh, cn);
        st_gelu(r.c3, mloc, bb, r0, nc + 48, rh, cn);
      }
      __syncthreads();

      // FFN pass 2: h = h + m @ W2^T + b2 (each wave reads only its own rows)
      for (int nc = 0; nc < DMODEL; nc += 64) {
        Acc4 r = gemm_strip4(mloc + r0 * DFF, DFF,
                             w2 + ((size_t)l * DMODEL + nc) * DFF, DFF, DFF);
        const float* bb = b2 + l * DMODEL;
        st_resid_bias(r.c0, hb, bb, r0, nc + 0,  rh, cn);
        st_resid_bias(r.c1, hb, bb, r0, nc + 16, rh, cn);
        st_resid_bias(r.c2, hb, bb, r0, nc + 32, rh, cn);
        st_resid_bias(r.c3, hb, bb, r0, nc + 48, rh, cn);
      }
      __syncthreads();
      layernorm_inplace(hb, ln2_g + l * DMODEL, ln2_b + l * DMODEL, tid);
      __syncthreads();
    }

    // -------- refine (tiny matrix-vector MLP, fp32 VALU) --------
    const int ptr = THIST + step;
    for (int i = tid; i < DMODEL; i += 256)
      comb[DMODEL + i] = (float)hb[(ptr - 1) * DMODEL + i];  // ctx (constant)
    if (tid < DIN) curv[tid] = buf[(ptr - 1) * DIN + tid];
    __syncthreads();
    for (int it = 0; it < NSUB; ++it) {
      for (int o = tid; o < DMODEL; o += 256) {
        float s = 0.f;
        for (int i = 0; i < DIN; ++i) s += W_emb_f[o * DIN + i] * curv[i];
        comb[o] = (s + b_emb[o]) * scale;
      }
      __syncthreads();
      for (int o = tid; o < DFF; o += 256) {
        float s = 0.f;
        for (int i = 0; i < 2 * DMODEL; ++i)
          s += Wr1[o * (2 * DMODEL) + i] * comb[i];
        actv[o] = gelu_tanh(s + br1[o]);
      }
      __syncthreads();
      if (tid < DIN) {
        float s = 0.f;
        for (int i = 0; i < DFF; ++i) s += Wr2[tid * DFF + i] * actv[i];
        curv[tid] += s + br2[tid];
      }
      __syncthreads();
    }
    if (tid < DIN) {
      const float v = curv[tid];
      buf[ptr * DIN + tid] = v;
      out[(b * NSTEPS + step) * DIN + tid] = v;
    }
    __syncthreads();
  }
}

// ---------------------------------------------------------------------------
// Host launcher
// ---------------------------------------------------------------------------
extern "C" void kernel_launch(void* const* d_in, const int* in_sizes, int n_in,
                              void* d_out, int out_size, void* d_ws, size_t ws_size,
                              hipStream_t stream) {
  (void)in_sizes; (void)n_in; (void)out_size; (void)ws_size;
  const float* history = (const float*)d_in[0];
  // d_in[1] = steps (fixed 8)
  const float* W_emb = (const float*)d_in[2];
  const float* b_emb = (const float*)d_in[3];
  const float* Wq    = (const float*)d_in[4];
  const float* Wk    = (const float*)d_in[5];
  const float* Wv    = (const float*)d_in[6];
  const float* Wo    = (const float*)d_in[7];
  const float* ln1_g = (const float*)d_in[8];
  const float* ln1_b = (const float*)d_in[9];
  const float* W1    = (const float*)d_in[10];
  const float* b1    = (const float*)d_in[11];
  const float* W2    = (const float*)d_in[12];
  const float* b2    = (const float*)d_in[13];
  const float* ln2_g = (const float*)d_in[14];
  const float* ln2_b = (const float*)d_in[15];
  const float* Wr1   = (const float*)d_in[16];
  const float* br1   = (const float*)d_in[17];
  const float* Wr2   = (const float*)d_in[18];
  const float* br2   = (const float*)d_in[19];
  float* out = (float*)d_out;

  char* ws = (char*)d_ws;
  size_t off = 0;
  auto take = [&](size_t bytes) -> char* {
    char* p = ws + off;
    off = (off + bytes + 255) & ~(size_t)255;
    return p;
  };
  const int nEmb = DMODEL * DIN;              // 16384
  const int nQ   = NLAYERS * DMODEL * DMODEL; // 1048576
  const int nF   = NLAYERS * DFF * DMODEL;    // 4194304
  bf16* wemb_bf = (bf16*)take((size_t)nEmb * 2);
  bf16* wq_bf   = (bf16*)take((size_t)nQ * 2);
  bf16* wk_bf   = (bf16*)take((size_t)nQ * 2);
  bf16* wv_bf   = (bf16*)take((size_t)nQ * 2);
  bf16* wo_bf   = (bf16*)take((size_t)nQ * 2);
  bf16* w1_bf   = (bf16*)take((size_t)nF * 2);
  bf16* w2_bf   = (bf16*)take((size_t)nF * 2);
  bf16* m_ws    = (bf16*)take((size_t)NBATCH * TSEQ * DFF * 2);
  float* buf_ws = (float*)take((size_t)NBATCH * TSEQ * DIN * 4);

  auto conv = [&](const float* s, bf16* d, int n) {
    int blocks = (n + 255) / 256;
    if (blocks > 2048) blocks = 2048;
    hipLaunchKernelGGL(f32_to_bf16_kernel, dim3(blocks), dim3(256), 0, stream,
                       s, d, n);
  };
  conv(W_emb, wemb_bf, nEmb);
  conv(Wq, wq_bf, nQ);
  conv(Wk, wk_bf, nQ);
  conv(Wv, wv_bf, nQ);
  conv(Wo, wo_bf, nQ);
  conv(W1, w1_bf, nF);
  conv(W2, w2_bf, nF);

  hipLaunchKernelGGL(rollout_kernel, dim3(NBATCH), dim3(256), 0, stream,
                     history, b_emb, ln1_g, ln1_b, b1, b2, ln2_g, ln2_b,
                     W_emb, Wr1, br1, Wr2, br2,
                     wemb_bf, wq_bf, wk_bf, wv_bf, wo_bf, w1_bf, w2_bf,
                     m_ws, buf_ws, out);
}
